// PatchFlow_40604620817220
// MI455X (gfx1250) — matrix operand order repaired and checked by reading, not compiled, not run on hardware
//
#include <hip/hip_runtime.h>
#include <math.h>

typedef __attribute__((ext_vector_type(16))) _Float16 v16h;
typedef __attribute__((ext_vector_type(8)))  _Float16 v8h;
typedef __attribute__((ext_vector_type(8)))  float    v8f;
typedef __attribute__((ext_vector_type(4)))  float    v4f;

#define PF_FLOWS 4
#define PF_P 4096          // 64*64 pixels
#define PF_B 16            // batch == rows per WMMA tile
#define PF_D 16
#define PF_HID 128
#define PF_PDIM 184
#define PF_PPAD 192
#define PF_TAIL 3.0f
#define PF_MINBIN 0.001f
#define PF_MINDER 0.001f
#define PF_INVSQRTHID 0.08838834764831843f
#define PF_LOG2PI 1.8378770664093453f

// per-wave LDS layout (bytes)
#define PW_ACT0 0                    // 16x128 f16
#define PW_ACT1 4096                 // 16x128 f16
#define PW_PRM  8192                 // 16x192 f32
#define PW_Z    (8192 + 12288)       // 16x16 f32
#define PW_BYTES (PW_Z + 1024)       // 21504
#define PF_WAVES 4

// ---------------- workspace map (bytes, 256-aligned) ----------------
#define WS_PATCH 0u
#define WS_T0    4194304u                        // patches: 65536*16*4
#define WS_GSIG  (WS_T0 + 8388608u)              // t0ctx: 4*4096*128*4
#define WS_LDET  (WS_GSIG + 16777216u)           // gsig: 8*4096*128*4
#define WS_FLU   (WS_LDET + 256u)
#define WS_FWZ   (WS_FLU + 4096u)                // 4 frags * 1KB
#define WS_FR1   (WS_FWZ + 32768u)               // 32 frags
#define WS_FR2   (WS_FR1 + 262144u)              // 256 frags
#define WS_FWF   (WS_FR2 + 262144u)              // 256 frags
// fragWf: 192 frags = 196608 B; total ws ~30.2 MB

__device__ __forceinline__ float pf_softplus(float x) {
  return (x > 20.f) ? x : __logf(1.f + __expf(x));
}
__device__ __forceinline__ v8f pf_splat8(float v) {
  v8f r = {v, v, v, v, v, v, v, v};
  return r;
}
// A-fragment (16x32 f16) from row-major LDS tile: contiguous-k A layout
// lane: row = lane&15, half h = lane>>4; element s: K = 16h+s (matches B fragments)
// -> K range [16h,16h+16) is contiguous: two 16B ds loads.
__device__ __forceinline__ v16h pf_load_a(const _Float16* tile, int lane, int kb, int rowStride) {
  int row = lane & 15, hh = lane >> 4;
  const _Float16* base = tile + row * rowStride + kb * 32 + 16 * hh;
  v8h lo = *(const v8h*)(base);
  v8h hi = *(const v8h*)(base + 8);
  return __builtin_shufflevector(lo, hi, 0, 1, 2, 3, 4, 5, 6, 7, 8, 9, 10, 11, 12, 13, 14, 15);
}
__device__ __forceinline__ v8f pf_wmma(v16h a, v16h b, v8f c) {
  return __builtin_amdgcn_wmma_f32_16x16x32_f16(false, a, false, b, (short)0, c, false, false);
}

// Forward rational-quadratic spline (nflows math), returns logabsdet term.
__device__ __forceinline__ float pf_rqspline(float x, const float* uw, const float* uh,
                                             const float* ud, float& yout) {
  bool inside = (x >= -PF_TAIL) && (x <= PF_TAIL);
  float xc = fminf(fmaxf(x, -PF_TAIL), PF_TAIL);
  float cw[9], chh[9], dv[9];
  // widths softmax -> cumwidths
  float mw = uw[0];
#pragma unroll
  for (int k = 1; k < 8; ++k) mw = fmaxf(mw, uw[k]);
  float ew[8], sw = 0.f;
#pragma unroll
  for (int k = 0; k < 8; ++k) { ew[k] = __expf(uw[k] - mw); sw += ew[k]; }
  float fw = (1.f - PF_MINBIN * 8.f) / sw;
  cw[0] = -PF_TAIL;
  float acc = 0.f;
#pragma unroll
  for (int k = 0; k < 8; ++k) { acc += PF_MINBIN + fw * ew[k]; cw[k + 1] = 2.f * PF_TAIL * acc - PF_TAIL; }
  cw[8] = PF_TAIL;
  // heights softmax -> cumheights
  float mh = uh[0];
#pragma unroll
  for (int k = 1; k < 8; ++k) mh = fmaxf(mh, uh[k]);
  float eh[8], sh = 0.f;
#pragma unroll
  for (int k = 0; k < 8; ++k) { eh[k] = __expf(uh[k] - mh); sh += eh[k]; }
  float fh = (1.f - PF_MINBIN * 8.f) / sh;
  chh[0] = -PF_TAIL;
  acc = 0.f;
#pragma unroll
  for (int k = 0; k < 8; ++k) { acc += PF_MINBIN + fh * eh[k]; chh[k + 1] = 2.f * PF_TAIL * acc - PF_TAIL; }
  chh[8] = PF_TAIL;
  // derivatives (boundary pad gives exactly 1.0)
  dv[0] = 1.f; dv[8] = 1.f;
#pragma unroll
  for (int k = 0; k < 7; ++k) dv[k + 1] = PF_MINDER + pf_softplus(ud[k]);
  // bin index
  int idx = 0;
#pragma unroll
  for (int j = 1; j < 9; ++j) idx += (xc >= cw[j]) ? 1 : 0;
  idx = (idx > 7) ? 7 : idx;
  // select-chain gather (keeps everything in registers)
  float icw = cw[0], iw = cw[1] - cw[0], ich = chh[0], ih = chh[1] - chh[0];
  float dk = dv[0], dk1 = dv[1];
#pragma unroll
  for (int k = 1; k < 8; ++k) {
    bool s = (idx == k);
    icw = s ? cw[k] : icw;
    iw = s ? (cw[k + 1] - cw[k]) : iw;
    ich = s ? chh[k] : ich;
    ih = s ? (chh[k + 1] - chh[k]) : ih;
    dk = s ? dv[k] : dk;
    dk1 = s ? dv[k + 1] : dk1;
  }
  float delta = ih / iw;
  float th = (xc - icw) / iw;
  float omt = 1.f - th;
  float t1m = th * omt;
  float denom = delta + (dk + dk1 - 2.f * delta) * t1m;
  float y = ich + ih * (delta * th * th + dk * t1m) / denom;
  float dnum = delta * delta * (dk1 * th * th + 2.f * delta * t1m + dk * omt * omt);
  float ld = __logf(dnum) - 2.f * __logf(denom);
  yout = inside ? y : x;
  return inside ? ld : 0.f;
}

// ================= kernel 0: pooled L2 patches =================
__global__ void patches_k(const float* __restrict__ x, float* __restrict__ patches) {
  int idx = blockIdx.x * 256 + threadIdx.x;       // 4096*256 total
  int d = idx & 15;
  int b = (idx >> 4) & 15;
  int p = idx >> 8;
  int h = p >> 6, w = p & 63;
  int y0 = 2 * h - 1, x0 = 2 * w - 1;
  float s = 0.f;
  for (int c = 0; c < 3; ++c) {
#pragma unroll
    for (int dy = 0; dy < 3; ++dy) {
      int yy = y0 + dy;
      if (yy < 0 || yy > 127) continue;
#pragma unroll
      for (int dx = 0; dx < 3; ++dx) {
        int xx = x0 + dx;
        if (xx < 0 || xx > 127) continue;
        float v = x[(((b * 16 + d) * 3 + c) * 128 + yy) * 128 + xx];
        s += v * v;
      }
    }
  }
  patches[idx] = sqrtf(s);   // layout [(p*16+b)*16 + d]
}

// ================= kernel 1: weight fragments (contiguous-k B layout) =================
// B layout used here (16-bit, 32x16, contiguous k): lane n = lane&15; lanes>=16 carry K 16..31; VGPR r: K = 2r(+16h)
__global__ void build_frags_k(const float* __restrict__ W_init, const float* __restrict__ W_res1,
                              const float* __restrict__ W_res2, const float* __restrict__ W_final,
                              const float* __restrict__ lu_lower, const float* __restrict__ lu_upper,
                              const float* __restrict__ lu_udiag, const int* __restrict__ perms,
                              v16h* fragLU, v16h* fragWz, v16h* fragR1, v16h* fragR2, v16h* fragWf,
                              float* logdet) {
  __shared__ float Lm[16][16], Um[16][16], LUm[16][16];
  __shared__ int inv[16];
  __shared__ float tmp[16];
  const int bid = blockIdx.x;
  const int lane = threadIdx.x;
  const int n = lane & 15;
  const int khalf = (lane >> 4) * 16;
  v16h v;

  if (bid < 4) {   // combined perm+LU matrix  z' = z @ (Pm U^T L^T)
    const int f = bid;
    for (int i = lane; i < 256; i += 32) {
      int r = i >> 4, c = i & 15;
      float lo = lu_lower[(f * 16 + r) * 16 + c];
      Lm[r][c] = (r == c) ? 1.f : ((r > c) ? lo : 0.f);
      float up = lu_upper[(f * 16 + r) * 16 + c];
      float dg = pf_softplus(lu_udiag[f * 16 + r]) + 0.001f;
      Um[r][c] = (r < c) ? up : ((r == c) ? dg : 0.f);
    }
    if (lane < 16) {
      int jj = 0;
      for (int j = 0; j < 16; ++j)
        if (perms[f * 16 + j] == lane) jj = j;
      inv[lane] = jj;
      tmp[lane] = __logf(pf_softplus(lu_udiag[f * 16 + lane]) + 0.001f);
    }
    __syncthreads();
    for (int i = lane; i < 256; i += 32) {
      int r = i >> 4, c = i & 15;
      float s = 0.f;
      for (int k = 0; k < 16; ++k) s += Lm[r][k] * Um[k][c];
      LUm[r][c] = s;
    }
    __syncthreads();
#pragma unroll
    for (int r = 0; r < 8; ++r)
#pragma unroll
      for (int t = 0; t < 2; ++t) {
        int k = khalf + 2 * r + t;
        v[2 * r + t] = (_Float16)((k < 16) ? LUm[n][inv[k]] : 0.f);
      }
    fragLU[f * 32 + lane] = v;
    if (lane == 0) {
      float s = 0.f;
      for (int dd = 0; dd < 16; ++dd) s += tmp[dd];
      logdet[f] = s;
    }
    return;
  }
  int id = bid - 4;
  if (id < 32) {   // Wz: idf part of W_init folded over full z (odd K -> 0)
    int f = id >> 3, nt = id & 7;
#pragma unroll
    for (int r = 0; r < 8; ++r)
#pragma unroll
      for (int t = 0; t < 2; ++t) {
        int k = khalf + 2 * r + t;
        float val = (k < 16 && ((k & 1) == 0)) ? W_init[(f * 128 + nt * 16 + n) * 136 + (k >> 1)] : 0.f;
        v[2 * r + t] = (_Float16)val;
      }
    fragWz[id * 32 + lane] = v;
    return;
  }
  id -= 32;
  if (id < 256) {  // res1
    int f = id >> 6, j = (id >> 5) & 1, nt = (id >> 2) & 7, kb = id & 3;
    const float* base = W_res1 + ((size_t)((f * 2 + j) * 128 + nt * 16 + n)) * 128 + kb * 32;
#pragma unroll
    for (int r = 0; r < 8; ++r)
#pragma unroll
      for (int t = 0; t < 2; ++t) v[2 * r + t] = (_Float16)base[khalf + 2 * r + t];
    fragR1[id * 32 + lane] = v;
    return;
  }
  id -= 256;
  if (id < 256) {  // res2
    int f = id >> 6, j = (id >> 5) & 1, nt = (id >> 2) & 7, kb = id & 3;
    const float* base = W_res2 + ((size_t)((f * 2 + j) * 128 + nt * 16 + n)) * 128 + kb * 32;
#pragma unroll
    for (int r = 0; r < 8; ++r)
#pragma unroll
      for (int t = 0; t < 2; ++t) v[2 * r + t] = (_Float16)base[khalf + 2 * r + t];
    fragR2[id * 32 + lane] = v;
    return;
  }
  id -= 256;
  {                // W_final with 1/sqrt(HID) folded into uw/uh columns, padded to 192
    int kb = id & 3, nt = (id >> 2) % 12, f = id / 48;
    int c = nt * 16 + n;
#pragma unroll
    for (int r = 0; r < 8; ++r)
#pragma unroll
      for (int t = 0; t < 2; ++t) {
        int k = khalf + 2 * r + t;
        float val = 0.f;
        if (c < PF_PDIM) {
          float sc = ((c % 23) < 16) ? PF_INVSQRTHID : 1.f;
          val = W_final[((size_t)f * PF_PDIM + c) * 128 + kb * 32 + k] * sc;
        }
        v[2 * r + t] = (_Float16)val;
      }
    fragWf[id * 32 + lane] = v;
  }
}

// ================= kernel 2: per-pixel ctx tables =================
__global__ void ctx_precompute_k(const float* __restrict__ W_init, const float* __restrict__ b_init,
                                 const float* __restrict__ W_ctx, const float* __restrict__ b_ctx,
                                 float* __restrict__ t0ctx, float* __restrict__ gsig) {
  __shared__ float ctx[128];
  int p = blockIdx.x;
  int t = threadIdx.x;                // 128
  int h = p >> 6, w = p & 63;
  {
    int half = t >> 6;                // 0: ex(h) ; 1: ey(w)
    int loc = t & 63;
    int q = loc >> 1;
    int pos = half ? w : h;
    float invf = __expf(-((float)q / 32.f) * __logf(10000.f));
    float s = (float)pos * invf;
    ctx[t] = (loc & 1) ? __cosf(s) : __sinf(s);
  }
  __syncthreads();
  for (int f = 0; f < PF_FLOWS; ++f) {
    float a = b_init[f * 128 + t];
    const float* wrow = W_init + ((size_t)(f * 128 + t)) * 136 + 8;
    for (int k = 0; k < 128; ++k) a += ctx[k] * wrow[k];
    t0ctx[((size_t)f * PF_P + p) * 128 + t] = a;
  }
  for (int fj = 0; fj < 8; ++fj) {
    float g = b_ctx[fj * 128 + t];
    const float* wrow = W_ctx + ((size_t)(fj * 128 + t)) * 128;
    for (int k = 0; k < 128; ++k) g += ctx[k] * wrow[k];
    gsig[((size_t)fj * PF_P + p) * 128 + t] = 1.f / (1.f + __expf(-g));
  }
}

// ================= kernel 3: main flow (1 wave = 1 pixel tile of 16 rows) =================
__global__ __launch_bounds__(128) void flow_main_k(
    const float* __restrict__ patches, const float* __restrict__ t0ctx,
    const float* __restrict__ gsig, const float* __restrict__ logdet,
    const v16h* __restrict__ fragLU, const v16h* __restrict__ fragWz,
    const v16h* __restrict__ fragR1, const v16h* __restrict__ fragR2,
    const v16h* __restrict__ fragWf,
    const float* __restrict__ b_res1, const float* __restrict__ b_res2,
    const float* __restrict__ b_final, const float* __restrict__ lu_bias,
    const float* __restrict__ uncond_w, const float* __restrict__ uncond_h,
    const float* __restrict__ uncond_d,
    const float* __restrict__ q0_loc, const float* __restrict__ q0_ls,
    float* __restrict__ out) {
  extern __shared__ char smem[];
  const int lane = threadIdx.x & 31;
  const int wv = threadIdx.x >> 5;
  const int p = blockIdx.x * PF_WAVES + wv;
  char* my = smem + (size_t)wv * PW_BYTES;
  _Float16* act0 = (_Float16*)(my + PW_ACT0);
  _Float16* act1 = (_Float16*)(my + PW_ACT1);
  float* prm = (float*)(my + PW_PRM);
  float* zf = (float*)(my + PW_Z);

  const int col = lane & 15;
  const int hh = lane >> 4;
  const int rbase = 8 * hh;
  const int row = lane & 15;

  // ---- init z tile from patches; z in f32 (zf) + f16 A-tile (act0, K 16..31 zeroed) ----
  {
    int dbase = 8 * hh;
    const float* src = patches + ((size_t)p * 16 + row) * 16 + dbase;
    v4f a0 = *(const v4f*)(src);
    v4f a1 = *(const v4f*)(src + 4);
#pragma unroll
    for (int i = 0; i < 4; ++i) {
      zf[row * 16 + dbase + i] = a0[i];
      zf[row * 16 + dbase + 4 + i] = a1[i];
      act0[row * 128 + dbase + i] = (_Float16)a0[i];
      act0[row * 128 + dbase + 4 + i] = (_Float16)a1[i];
    }
#pragma unroll
    for (int i = 0; i < 8; ++i) act0[row * 128 + 16 + 8 * hh + i] = (_Float16)0.f;
  }
  __syncthreads();

  float lq = 0.f;   // meaningful on lanes 0..15 (per-row log_q)

#pragma unroll 1
  for (int f = PF_FLOWS - 1; f >= 0; --f) {
    // ---- perm + LU:  z' = z @ (Pm U^T L^T) + lu_bias ----
    {
      v16h a = pf_load_a(act0, lane, 0, 128);
      v16h b = fragLU[f * 32 + lane];
      v8f c = pf_splat8(lu_bias[f * 16 + col]);
      c = pf_wmma(a, b, c);
      __syncthreads();
#pragma unroll
      for (int r = 0; r < 8; ++r) {
        float vv = c[r];
        int rw = rbase + r;
        zf[rw * 16 + col] = vv;
        act0[rw * 128 + col] = (_Float16)vv;
      }
    }
    __syncthreads();

    // ---- t = [idf|ctx] @ W_init^T + b_init  (ctx part precomputed) ----
    v8f tf[8];
    {
      v16h az = pf_load_a(act0, lane, 0, 128);
#pragma unroll
      for (int nt = 0; nt < 8; ++nt) {
        v8f acc = pf_splat8(t0ctx[((size_t)f * PF_P + p) * 128 + nt * 16 + col]);
        v16h b = fragWz[(f * 8 + nt) * 32 + lane];
        tf[nt] = pf_wmma(az, b, acc);
      }
    }

    // ---- two gated residual blocks ----
#pragma unroll 1
    for (int j = 0; j < 2; ++j) {
#pragma unroll
      for (int nt = 0; nt < 8; ++nt)
#pragma unroll
        for (int r = 0; r < 8; ++r)
          act0[(rbase + r) * 128 + nt * 16 + col] = (_Float16)fmaxf(tf[nt][r], 0.f);
      __syncthreads();
#pragma unroll
      for (int nt = 0; nt < 8; ++nt) {
        v8f acc = pf_splat8(b_res1[(f * 2 + j) * 128 + nt * 16 + col]);
#pragma unroll
        for (int kb = 0; kb < 4; ++kb) {
          v16h a = pf_load_a(act0, lane, kb, 128);
          v16h b = fragR1[(f * 64 + j * 32 + nt * 4 + kb) * 32 + lane];
          acc = pf_wmma(a, b, acc);
        }
#pragma unroll
        for (int r = 0; r < 8; ++r)
          act1[(rbase + r) * 128 + nt * 16 + col] = (_Float16)fmaxf(acc[r], 0.f);
      }
      __syncthreads();
#pragma unroll
      for (int nt = 0; nt < 8; ++nt) {
        v8f acc = pf_splat8(b_res2[(f * 2 + j) * 128 + nt * 16 + col]);
#pragma unroll
        for (int kb = 0; kb < 4; ++kb) {
          v16h a = pf_load_a(act1, lane, kb, 128);
          v16h b = fragR2[(f * 64 + j * 32 + nt * 4 + kb) * 32 + lane];
          acc = pf_wmma(a, b, acc);
        }
        float g = gsig[((size_t)(f * 2 + j) * PF_P + p) * 128 + nt * 16 + col];
#pragma unroll
        for (int r = 0; r < 8; ++r) tf[nt][r] += acc[r] * g;
      }
      __syncthreads();
    }

    // ---- params = t @ W_final^T + b_final (scaled) ----
#pragma unroll
    for (int nt = 0; nt < 8; ++nt)
#pragma unroll
      for (int r = 0; r < 8; ++r)
        act0[(rbase + r) * 128 + nt * 16 + col] = (_Float16)tf[nt][r];
    __syncthreads();
#pragma unroll
    for (int nt = 0; nt < 12; ++nt) {
      int cidx = nt * 16 + col;
      float bb = 0.f;
      if (cidx < PF_PDIM) {
        float sc = ((cidx % 23) < 16) ? PF_INVSQRTHID : 1.f;
        bb = b_final[f * PF_PDIM + cidx] * sc;
      }
      v8f acc = pf_splat8(bb);
#pragma unroll
      for (int kb = 0; kb < 4; ++kb) {
        v16h a = pf_load_a(act0, lane, kb, 128);
        v16h b = fragWf[(f * 48 + nt * 4 + kb) * 32 + lane];
        acc = pf_wmma(a, b, acc);
      }
#pragma unroll
      for (int r = 0; r < 8; ++r) prm[(rbase + r) * PF_PPAD + cidx] = acc[r];
    }
    __syncthreads();

    // ---- spline phase: lane handles its row's dims [4h..4h+3] ----
    float ldsum = 0.f;
    float zid[4], ztr[4];
#pragma unroll 1
    for (int mi = 0; mi < 4; ++mi) {
      int m = hh * 4 + mi;
      float uw[8], uh8[8], ud7[7];
      const float* pp = prm + row * PF_PPAD + m * 23;
#pragma unroll
      for (int k = 0; k < 8; ++k) { uw[k] = pp[k]; uh8[k] = pp[8 + k]; }
#pragma unroll
      for (int k = 0; k < 7; ++k) ud7[k] = pp[16 + k];
      float y;
      ldsum += pf_rqspline(zf[row * 16 + 2 * m + 1], uw, uh8, ud7, y);
      ztr[mi] = y;
#pragma unroll
      for (int k = 0; k < 8; ++k) {
        uw[k] = uncond_w[(f * 8 + m) * 8 + k];
        uh8[k] = uncond_h[(f * 8 + m) * 8 + k];
      }
#pragma unroll
      for (int k = 0; k < 7; ++k) ud7[k] = uncond_d[(f * 8 + m) * 7 + k];
      ldsum += pf_rqspline(zf[row * 16 + 2 * m], uw, uh8, ud7, y);
      zid[mi] = y;
    }
    ldsum += __shfl_xor(ldsum, 16);
    if (lane < 16) lq += ldsum + logdet[f];
    __syncthreads();
    // ---- rebuild z for next flow (interleave id/tr), refresh f16 A-tile ----
#pragma unroll
    for (int mi = 0; mi < 4; ++mi) {
      int m = hh * 4 + mi;
      zf[row * 16 + 2 * m] = zid[mi];
      zf[row * 16 + 2 * m + 1] = ztr[mi];
      act0[row * 128 + 2 * m] = (_Float16)zid[mi];
      act0[row * 128 + 2 * m + 1] = (_Float16)ztr[mi];
    }
#pragma unroll
    for (int i = 0; i < 8; ++i) act0[row * 128 + 16 + 8 * hh + i] = (_Float16)0.f;
    __syncthreads();
  }

  // ---- Gaussian base density + transposed store out[b,h,w] ----
  if (lane < 16) {
    float s = 0.f;
#pragma unroll
    for (int d = 0; d < 16; ++d) {
      float ls = q0_ls[d];
      float u = (zf[row * 16 + d] - q0_loc[d]) * __expf(-ls);
      s += ls + 0.5f * u * u;
    }
    out[(size_t)row * PF_P + p] = lq - 8.f * PF_LOG2PI - s;
  }
}

extern "C" void kernel_launch(void* const* d_in, const int* in_sizes, int n_in,
                              void* d_out, int out_size, void* d_ws, size_t ws_size,
                              hipStream_t stream) {
  (void)in_sizes; (void)n_in; (void)out_size; (void)ws_size;
  const float* x        = (const float*)d_in[0];
  const float* W_init   = (const float*)d_in[1];
  const float* b_init   = (const float*)d_in[2];
  const float* W_ctx    = (const float*)d_in[3];
  const float* b_ctx    = (const float*)d_in[4];
  const float* W_res1   = (const float*)d_in[5];
  const float* b_res1   = (const float*)d_in[6];
  const float* W_res2   = (const float*)d_in[7];
  const float* b_res2   = (const float*)d_in[8];
  const float* W_final  = (const float*)d_in[9];
  const float* b_final  = (const float*)d_in[10];
  const float* uncond_w = (const float*)d_in[11];
  const float* uncond_h = (const float*)d_in[12];
  const float* uncond_d = (const float*)d_in[13];
  const float* lu_lower = (const float*)d_in[14];
  const float* lu_upper = (const float*)d_in[15];
  const float* lu_udiag = (const float*)d_in[16];
  const float* lu_bias  = (const float*)d_in[17];
  const float* q0_loc   = (const float*)d_in[18];
  const float* q0_ls    = (const float*)d_in[19];
  const int*   perms    = (const int*)d_in[20];

  char* ws = (char*)d_ws;
  float* patches = (float*)(ws + WS_PATCH);
  float* t0ctx   = (float*)(ws + WS_T0);
  float* gsig    = (float*)(ws + WS_GSIG);
  float* logdet  = (float*)(ws + WS_LDET);
  v16h* fragLU   = (v16h*)(ws + WS_FLU);
  v16h* fragWz   = (v16h*)(ws + WS_FWZ);
  v16h* fragR1   = (v16h*)(ws + WS_FR1);
  v16h* fragR2   = (v16h*)(ws + WS_FR2);
  v16h* fragWf   = (v16h*)(ws + WS_FWF);

  patches_k<<<4096, 256, 0, stream>>>(x, patches);
  build_frags_k<<<740, 32, 0, stream>>>(W_init, W_res1, W_res2, W_final, lu_lower, lu_upper,
                                        lu_udiag, perms, fragLU, fragWz, fragR1, fragR2, fragWf,
                                        logdet);
  ctx_precompute_k<<<PF_P, 128, 0, stream>>>(W_init, b_init, W_ctx, b_ctx, t0ctx, gsig);
  flow_main_k<<<PF_P / PF_WAVES, 32 * PF_WAVES, PW_BYTES * PF_WAVES, stream>>>(
      patches, t0ctx, gsig, logdet, fragLU, fragWz, fragR1, fragR2, fragWf,
      b_res1, b_res2, b_final, lu_bias, uncond_w, uncond_h, uncond_d,
      q0_loc, q0_ls, (float*)d_out);
}
